// Network_13159779795277
// MI455X (gfx1250) — compile-verified, matrix operand order checked by reading, not run on hardware
//
#include <hip/hip_runtime.h>
#include <hip/hip_bf16.h>

typedef __attribute__((ext_vector_type(16))) _Float16 v16h;
typedef __attribute__((ext_vector_type(8)))  _Float16 v8h;
typedef __attribute__((ext_vector_type(8)))  float    v8f;

union AU { v16h v; v8h h[2]; };

#define KTAPS 27
#define CCH   64
// swizzled weight halves per conv layer: 27 * 4 nblk * 2 kc * 32 lanes * 16 halves
#define WLAYER 110592

#define WMMA_F16(A, B, C) __builtin_amdgcn_wmma_f32_16x16x32_f16( \
    false, (A), false, (B), (short)0, (C), false, false)

// ---------------------------------------------------------------------------
// Weight conversion: f32 (L,27,64,64) row-major [k][K][n] -> f16 swizzled into
// WMMA B-fragment order: [k][nblk][kc][lane][h], where lane%16 = N column,
// K = kc*32 + (lane>=16 ? 8 : 0) + (h<8 ? h : h+8)   (matches ISA 16-bit layout)
// ---------------------------------------------------------------------------
__global__ void cvt_wsw_kernel(const float* __restrict__ src,
                               _Float16* __restrict__ dst, int total) {
    int d = blockIdx.x * blockDim.x + threadIdx.x;
    if (d >= total) return;
    int layer = d / WLAYER;
    int r  = d % WLAYER;
    int k  = r >> 12;
    int r2 = r & 4095;
    int nb = r2 >> 10;
    int r3 = r2 & 1023;
    int kc = r3 >> 9;
    int r4 = r3 & 511;
    int lane = r4 >> 4;
    int h    = r4 & 15;
    int n  = nb * 16 + (lane & 15);
    int kb = kc * 32 + ((lane >> 4) << 3);
    int K  = kb + (h < 8 ? h : h + 8);
    dst[d] = (_Float16)src[(((size_t)layer * KTAPS + k) * CCH + K) * CCH + n];
}

__global__ void embed_l1_kernel(const float* __restrict__ pe,
                                const int* __restrict__ occ,
                                _Float16* __restrict__ out, int N) {
    int g = blockIdx.x * blockDim.x + threadIdx.x;
    if (g >= N * CCH) return;
    int i = g >> 6, c = g & 63;
    out[g] = (_Float16)pe[occ[i] * CCH + c];
}

__global__ void embed_l2_kernel(const _Float16* __restrict__ fpar,
                                const int* __restrict__ parent,
                                const int* __restrict__ oct,
                                const float* __restrict__ emb8,
                                _Float16* __restrict__ out, int N) {
    int g = blockIdx.x * blockDim.x + threadIdx.x;
    if (g >= N * CCH) return;
    int i = g >> 6, c = g & 63;
    float v = (float)fpar[(size_t)parent[i] * CCH + c] + emb8[oct[i] * CCH + c];
    out[g] = (_Float16)v;
}

// ---------------------------------------------------------------------------
// Sparse 27-tap conv via WMMA. out = relu( (resid?resid:0) + sum_k gather_k @ W[k] )
// 256 threads = 8 wave32; each wave owns 32 rows x 64 cols (two 16-row M-tiles)
// so each LDS B-fragment read feeds two WMMAs. Weight slices are staged into a
// double-buffered LDS region with GLOBAL_LOAD_ASYNC_TO_LDS_B128 (ASYNCcnt).
// ---------------------------------------------------------------------------
__global__ void __launch_bounds__(256)
sp_conv_wmma_kernel(const _Float16* __restrict__ fin,
                    const int* __restrict__ nbr,
                    const _Float16* __restrict__ wsw,   // one layer, WLAYER halves
                    const _Float16* __restrict__ resid, // nullable
                    _Float16* __restrict__ fout, int N) {
    __shared__ __align__(16) _Float16 sB[2][4096];   // double-buffered 64x64 f16 slice

    const int tid  = threadIdx.x;
    const int lane = tid & 31;
    const int wave = tid >> 5;
    const int row0 = (blockIdx.x * 8 + wave) * 32;
    const int m    = lane & 15;
    const int hiK  = (lane >> 4) << 3;   // 0 for lanes 0-15, 8 for lanes 16-31

    const int rowA = row0 + m;
    const int rowB = row0 + 16 + m;
    const bool rvA = rowA < N;
    const bool rvB = rowB < N;

    // LDS destination addresses (32B per thread per buffer) for async staging
    const unsigned lbase0 = (unsigned)(size_t)&sB[0][tid * 16];
    const unsigned lbase1 = (unsigned)(size_t)&sB[1][tid * 16];
    const unsigned voff   = (unsigned)(tid * 32);   // global byte offset

    // Prologue: stage tap 0 into buffer 0 (2 x b128 per thread; INST_OFFSET
    // advances both the global source and the LDS destination).
    {
        uint64_t g = (uint64_t)(size_t)wsw;
        asm volatile("global_load_async_to_lds_b128 %0, %1, %2\n\t"
                     "global_load_async_to_lds_b128 %0, %1, %2 offset:16"
                     :: "v"(lbase0), "v"(voff), "s"(g) : "memory");
    }

    v8f acc[2][4] = {};

    for (int k = 0; k < KTAPS; ++k) {
        asm volatile("s_wait_asynccnt 0x0" ::: "memory");
        __syncthreads();   // buffer (k&1) fully staged for all waves

        if (k + 1 < KTAPS) {   // kick off next tap's weights into other buffer
            uint64_t g = (uint64_t)(size_t)(wsw + (size_t)(k + 1) * 4096);
            unsigned lb = ((k + 1) & 1) ? lbase1 : lbase0;
            asm volatile("global_load_async_to_lds_b128 %0, %1, %2\n\t"
                         "global_load_async_to_lds_b128 %0, %1, %2 offset:16"
                         :: "v"(lb), "v"(voff), "s"(g) : "memory");
        }

        const _Float16* bufp = sB[k & 1];

        int idxA = rvA ? nbr[rowA * KTAPS + k] : -1;
        int idxB = rvB ? nbr[rowB * KTAPS + k] : -1;

        AU a00 = {}, a01 = {}, a10 = {}, a11 = {};
        if (idxA >= 0) {
            const _Float16* fr = fin + (size_t)idxA * CCH + hiK;
            a00.h[0] = *(const v8h*)(fr);        // kc=0, K = hiK+0..7
            a00.h[1] = *(const v8h*)(fr + 16);   // kc=0, K = 16+hiK+0..7
            a01.h[0] = *(const v8h*)(fr + 32);   // kc=1
            a01.h[1] = *(const v8h*)(fr + 48);
        }
        if (idxB >= 0) {
            const _Float16* fr = fin + (size_t)idxB * CCH + hiK;
            a10.h[0] = *(const v8h*)(fr);
            a10.h[1] = *(const v8h*)(fr + 16);
            a11.h[0] = *(const v8h*)(fr + 32);
            a11.h[1] = *(const v8h*)(fr + 48);
        }

        #pragma unroll
        for (int nb = 0; nb < 4; ++nb) {
            AU b0, b1;
            const v8h* bp0 = (const v8h*)&bufp[nb * 1024 + lane * 16];
            const v8h* bp1 = (const v8h*)&bufp[nb * 1024 + 512 + lane * 16];
            b0.h[0] = bp0[0]; b0.h[1] = bp0[1];
            b1.h[0] = bp1[0]; b1.h[1] = bp1[1];
            acc[0][nb] = WMMA_F16(a00.v, b0.v, acc[0][nb]);
            acc[0][nb] = WMMA_F16(a01.v, b1.v, acc[0][nb]);
            acc[1][nb] = WMMA_F16(a10.v, b0.v, acc[1][nb]);
            acc[1][nb] = WMMA_F16(a11.v, b1.v, acc[1][nb]);
        }
        __syncthreads();   // protect buffer overwritten two taps later
    }

    // Epilogue: C layout -> VGPR j, lanes 0-15: M=j, lanes 16-31: M=8+j, N=lane%16
    const int col = lane & 15;
    #pragma unroll
    for (int t = 0; t < 2; ++t) {
        const int rbase = row0 + t * 16 + hiK;
        #pragma unroll
        for (int nb = 0; nb < 4; ++nb) {
            #pragma unroll
            for (int j = 0; j < 8; ++j) {
                int r = rbase + j;
                if (r < N) {
                    float v = acc[t][nb][j];
                    size_t o = (size_t)r * CCH + nb * 16 + col;
                    if (resid) v += (float)resid[o];
                    v = fmaxf(v, 0.0f);
                    fout[o] = (_Float16)v;
                }
            }
        }
    }
}

// ---------------------------------------------------------------------------
// Softmax heads + bits accumulation. 16 threads per point, 16 points per block.
// ---------------------------------------------------------------------------
__global__ void __launch_bounds__(256)
heads_kernel(const _Float16* __restrict__ g,
             const float* __restrict__ W0a, const float* __restrict__ b0a,
             const float* __restrict__ W0b, const float* __restrict__ b0b,
             const float* __restrict__ W1a, const float* __restrict__ b1a,
             const float* __restrict__ W1b, const float* __restrict__ b1b,
             const float* __restrict__ s1,  const int* __restrict__ occ,
             float* __restrict__ accum, int N) {
    __shared__ float sg[16][64];
    __shared__ float sh[16][64];
    __shared__ float sl[16][16];
    __shared__ float sBits;

    const int tid = threadIdx.x;
    const int pl  = tid >> 4;
    const int t   = tid & 15;
    const int p   = blockIdx.x * 16 + pl;
    const bool valid = p < N;

    if (tid == 0) sBits = 0.0f;
    #pragma unroll
    for (int j = 0; j < 4; ++j) {
        int c = t + 16 * j;
        sg[pl][c] = valid ? (float)g[(size_t)p * CCH + c] : 0.0f;
    }
    __syncthreads();

    // head 0: hidden = relu(g @ W0a + b0a)
    for (int j = 0; j < 4; ++j) {
        int c = t + 16 * j;
        float a = b0a[c];
        for (int i = 0; i < 64; ++i) a += sg[pl][i] * W0a[i * 64 + c];
        sh[pl][c] = fmaxf(a, 0.0f);
    }
    __syncthreads();
    {
        float a = b0b[t];
        for (int i = 0; i < 64; ++i) a += sh[pl][i] * W0b[i * 16 + t];
        sl[pl][t] = a;
    }
    __syncthreads();

    int o = valid ? occ[p] : 0;
    int lower = o & 15;
    int upper = o >> 4;
    float bits = 0.0f;
    if (t == 0 && valid) {
        float mx = sl[pl][0];
        for (int i = 1; i < 16; ++i) mx = fmaxf(mx, sl[pl][i]);
        float s = 0.0f;
        for (int i = 0; i < 16; ++i) s += __expf(sl[pl][i] - mx);
        float pgt = __expf(sl[pl][lower] - mx) / s;
        bits += fminf(50.0f, fmaxf(0.0f, -__log2f(pgt + 1e-10f)));
    }

    // head 1: h1 = g + s1_emb[lower]
    for (int j = 0; j < 4; ++j) {
        int c = t + 16 * j;
        float a = b1a[c];
        for (int i = 0; i < 64; ++i)
            a += (sg[pl][i] + s1[lower * 64 + i]) * W1a[i * 64 + c];
        sh[pl][c] = fmaxf(a, 0.0f);
    }
    __syncthreads();
    {
        float a = b1b[t];
        for (int i = 0; i < 64; ++i) a += sh[pl][i] * W1b[i * 16 + t];
        sl[pl][t] = a;
    }
    __syncthreads();

    if (t == 0 && valid) {
        float mx = sl[pl][0];
        for (int i = 1; i < 16; ++i) mx = fmaxf(mx, sl[pl][i]);
        float s = 0.0f;
        for (int i = 0; i < 16; ++i) s += __expf(sl[pl][i] - mx);
        float pgt = __expf(sl[pl][upper] - mx) / s;
        bits += fminf(50.0f, fmaxf(0.0f, -__log2f(pgt + 1e-10f)));
        atomicAdd(&sBits, bits);
    }
    __syncthreads();
    if (tid == 0) atomicAdd(accum, sBits);
}

__global__ void init_accum_kernel(float* a) { a[0] = 0.0f; }

__global__ void finalize_kernel(const float* a, const int* np, float* out) {
    out[0] = a[0] / (float)np[0];
}

// ---------------------------------------------------------------------------
extern "C" void kernel_launch(void* const* d_in, const int* in_sizes, int n_in,
                              void* d_out, int out_size, void* d_ws, size_t ws_size,
                              hipStream_t stream) {
    const float* prior_emb   = (const float*)d_in[0];
    const float* target_emb8 = (const float*)d_in[1];
    const float* Wp          = (const float*)d_in[2];
    const float* Wt          = (const float*)d_in[3];
    const float* W0a = (const float*)d_in[4];
    const float* b0a = (const float*)d_in[5];
    const float* W0b = (const float*)d_in[6];
    const float* b0b = (const float*)d_in[7];
    const float* W1a = (const float*)d_in[8];
    const float* b1a = (const float*)d_in[9];
    const float* W1b = (const float*)d_in[10];
    const float* b1b = (const float*)d_in[11];
    const float* s1_emb     = (const float*)d_in[12];
    const int* curr_occ     = (const int*)d_in[13];
    const int* next_occ     = (const int*)d_in[14];
    const int* next_octant  = (const int*)d_in[15];
    const int* parent_idx   = (const int*)d_in[16];
    const int* nbr_curr     = (const int*)d_in[17];
    const int* nbr_next     = (const int*)d_in[18];
    const int* n_points_dev = (const int*)d_in[19];

    const int N1 = in_sizes[13];        // curr_occ is (N1,1)
    const int N2 = in_sizes[16];        // parent_idx is (N2,)

    const size_t WHALF = (size_t)5 * WLAYER;       // halves per weight tensor
    const size_t FELEM = (size_t)N2 * CCH;         // halves per feature buffer

    char* ws = (char*)d_ws;
    float*    accum = (float*)ws;
    _Float16* Wp16  = (_Float16*)(ws + 64);
    _Float16* Wt16  = Wp16 + WHALF;
    _Float16* fA    = Wt16 + WHALF;
    _Float16* fB    = fA + FELEM;
    _Float16* fC    = fB + FELEM;

    init_accum_kernel<<<1, 1, 0, stream>>>(accum);

    {   // weight conversion (both conv stacks)
        int total = (int)WHALF;
        int blks = (total + 255) / 256;
        cvt_wsw_kernel<<<blks, 256, 0, stream>>>(Wp, Wp16, total);
        cvt_wsw_kernel<<<blks, 256, 0, stream>>>(Wt, Wt16, total);
    }

    // ---- level 1 ----
    embed_l1_kernel<<<(N1 * CCH + 255) / 256, 256, 0, stream>>>(prior_emb, curr_occ, fA, N1);

    dim3 cgrid1((N1 + 255) / 256);
    sp_conv_wmma_kernel<<<cgrid1, 256, 0, stream>>>(fA, nbr_curr, Wp16 + 0 * WLAYER, nullptr, fB, N1);
    sp_conv_wmma_kernel<<<cgrid1, 256, 0, stream>>>(fB, nbr_curr, Wp16 + 1 * WLAYER, nullptr, fC, N1);
    sp_conv_wmma_kernel<<<cgrid1, 256, 0, stream>>>(fC, nbr_curr, Wp16 + 2 * WLAYER, fB,      fA, N1);
    sp_conv_wmma_kernel<<<cgrid1, 256, 0, stream>>>(fA, nbr_curr, Wp16 + 3 * WLAYER, nullptr, fB, N1);
    sp_conv_wmma_kernel<<<cgrid1, 256, 0, stream>>>(fB, nbr_curr, Wp16 + 4 * WLAYER, fA,      fC, N1);
    // level-1 final features in fC

    // ---- level 2 ----
    embed_l2_kernel<<<(N2 * CCH + 255) / 256, 256, 0, stream>>>(fC, parent_idx, next_octant,
                                                                target_emb8, fA, N2);
    dim3 cgrid2((N2 + 255) / 256);
    sp_conv_wmma_kernel<<<cgrid2, 256, 0, stream>>>(fA, nbr_next, Wt16 + 0 * WLAYER, nullptr, fB, N2);
    sp_conv_wmma_kernel<<<cgrid2, 256, 0, stream>>>(fB, nbr_next, Wt16 + 1 * WLAYER, nullptr, fC, N2);
    sp_conv_wmma_kernel<<<cgrid2, 256, 0, stream>>>(fC, nbr_next, Wt16 + 2 * WLAYER, fB,      fA, N2);
    sp_conv_wmma_kernel<<<cgrid2, 256, 0, stream>>>(fA, nbr_next, Wt16 + 3 * WLAYER, nullptr, fB, N2);
    sp_conv_wmma_kernel<<<cgrid2, 256, 0, stream>>>(fB, nbr_next, Wt16 + 4 * WLAYER, fA,      fC, N2);
    // level-2 final features in fC

    // ---- heads + bits ----
    heads_kernel<<<(N2 + 15) / 16, 256, 0, stream>>>(fC, W0a, b0a, W0b, b0b,
                                                     W1a, b1a, W1b, b1b,
                                                     s1_emb, next_occ, accum, N2);

    finalize_kernel<<<1, 1, 0, stream>>>(accum, n_points_dev, (float*)d_out);
}
